// SR_Matcher_4209067950291
// MI455X (gfx1250) — compile-verified, matrix operand order checked by reading, not run on hardware
//
#include <hip/hip_runtime.h>

// Problem dims
#define Bz 32
#define Sz 1024
#define Vz 300
#define PRE 300
#define FLG 16
#define RD  200
#define Mz  (Bz * Sz)        // 32768 rows

// Padded dims (multiples of 16 / 32 so WMMA loops need no masks)
#define K1P 320              // 316 -> 320 (10 k-steps of 32)
#define N1P 224              // 200 -> 224 (14 N-tiles)  == K of GEMM2
#define K2P 224              // 7 k-steps of 32
#define N2P 304              // 300 -> 304 (19 N-tiles)

typedef __attribute__((ext_vector_type(16))) __bf16 v16bf;
typedef __attribute__((ext_vector_type(8)))  float  v8f;

union FragAB { uint4 q[2]; unsigned u[8]; v16bf v; };
union FragC  { v8f v; float f[8]; };

__device__ __forceinline__ unsigned short f2bf_rne(float f) {
  unsigned x = __float_as_uint(f);
  return (unsigned short)((x + 0x7FFFu + ((x >> 16) & 1u)) >> 16);
}
__device__ __forceinline__ float bf2f(unsigned short h) {
  return __uint_as_float((unsigned)h << 16);
}
__device__ __forceinline__ void split_bf16(float f, unsigned short& h, unsigned short& l) {
  h = f2bf_rne(f);
  l = f2bf_rne(f - bf2f(h));
}

// ---------------- prep: matrix[316,200] -> matT hi/lo [224,320] (zero padded)
__global__ void prep_matT(const float* __restrict__ matrix,
                          unsigned short* __restrict__ mH,
                          unsigned short* __restrict__ mL) {
  int idx = blockIdx.x * 256 + threadIdx.x;
  if (idx >= N1P * K1P) return;
  int n = idx / K1P, k = idx - n * K1P;
  float v = (n < RD && k < (PRE + FLG)) ? matrix[k * RD + n] : 0.0f;
  unsigned short h, l; split_bf16(v, h, l);
  mH[idx] = h; mL[idx] = l;
}

// ---------------- prep: role_vectors[32, 300*200] -> roles hi/lo [32,304,224]
__global__ void prep_roles(const float* __restrict__ role,
                           unsigned short* __restrict__ rH,
                           unsigned short* __restrict__ rL) {
  int idx = blockIdx.x * 256 + threadIdx.x;
  if (idx >= Bz * N2P * K2P) return;
  int b = idx / (N2P * K2P);
  int r = idx - b * (N2P * K2P);
  int n = r / K2P, k = r - n * K2P;
  float v = (n < Vz && k < RD) ? role[(size_t)b * (Vz * RD) + n * RD + k] : 0.0f;
  unsigned short h, l; split_bf16(v, h, l);
  rH[idx] = h; rL[idx] = l;
}

// ---------------- GEMM1: proj = concat(pre,wid) @ matrix   (C-stationary, 14 N-tiles)
__global__ void __launch_bounds__(256)
gemm1_bf16x3(const float* __restrict__ pre, const float* __restrict__ wid,
             const unsigned short* __restrict__ mTH, const unsigned short* __restrict__ mTL,
             unsigned short* __restrict__ projH, unsigned short* __restrict__ projL) {
  const int lane = threadIdx.x & 31;
  const int wave = threadIdx.x >> 5;
  const int l16  = lane & 15;
  const int half = lane >> 4;                 // 0: K 0..7 / 16..23 ; 1: K 8..15 / 24..31
  const int Mtile = blockIdx.x * 8 + wave;    // 0..2047
  const int mrow  = Mtile * 16 + l16;

  FragC c[14];
  #pragma unroll
  for (int nt = 0; nt < 14; ++nt)
    #pragma unroll
    for (int r = 0; r < 8; ++r) c[nt].f[r] = 0.0f;

  const float* preRow = pre + (size_t)mrow * PRE;
  const float* widRow = wid + (size_t)mrow * FLG;

  for (int ks = 0; ks < 10; ++ks) {
    const int Kb = ks * 32 + half * 8;
    float q[16];
    if (ks < 9) {                              // fully inside pretrained (K<=287<300)
      __builtin_prefetch(preRow + Kb + 32, 0, 0);   // global_prefetch_b8: next k-step
      float4 f0 = *(const float4*)(preRow + Kb);
      float4 f1 = *(const float4*)(preRow + Kb + 4);
      float4 f2 = *(const float4*)(preRow + Kb + 16);
      float4 f3 = *(const float4*)(preRow + Kb + 20);
      q[0]=f0.x; q[1]=f0.y; q[2]=f0.z; q[3]=f0.w;
      q[4]=f1.x; q[5]=f1.y; q[6]=f1.z; q[7]=f1.w;
      q[8]=f2.x; q[9]=f2.y; q[10]=f2.z; q[11]=f2.w;
      q[12]=f3.x; q[13]=f3.y; q[14]=f3.z; q[15]=f3.w;
    } else {                                   // K 288..319: pretrained | word_id | zero pad
      #pragma unroll
      for (int j = 0; j < 8; ++j) {
        int K = Kb + j;
        q[j] = (K < PRE) ? preRow[K] : ((K < PRE + FLG) ? widRow[K - PRE] : 0.0f);
      }
      #pragma unroll
      for (int j = 0; j < 8; ++j) {
        int K = Kb + 16 + j;
        q[8 + j] = (K < PRE) ? preRow[K] : ((K < PRE + FLG) ? widRow[K - PRE] : 0.0f);
      }
    }
    FragAB aH, aL;
    #pragma unroll
    for (int r = 0; r < 8; ++r) {              // reg r holds K-pair (q[2r], q[2r+1])
      unsigned short h0, l0, h1, l1;
      split_bf16(q[2 * r], h0, l0);
      split_bf16(q[2 * r + 1], h1, l1);
      aH.u[r] = (unsigned)h0 | ((unsigned)h1 << 16);
      aL.u[r] = (unsigned)l0 | ((unsigned)l1 << 16);
    }
    #pragma unroll
    for (int nt = 0; nt < 14; ++nt) {
      const unsigned short* bH_p = mTH + (size_t)(nt * 16 + l16) * K1P + ks * 32 + half * 8;
      const unsigned short* bL_p = mTL + (size_t)(nt * 16 + l16) * K1P + ks * 32 + half * 8;
      FragAB bH, bL;
      bH.q[0] = *(const uint4*)(bH_p);
      bH.q[1] = *(const uint4*)(bH_p + 16);
      bL.q[0] = *(const uint4*)(bL_p);
      bL.q[1] = *(const uint4*)(bL_p + 16);
      c[nt].v = __builtin_amdgcn_wmma_f32_16x16x32_bf16(false, aH.v, false, bH.v, (short)0, c[nt].v, false, false);
      c[nt].v = __builtin_amdgcn_wmma_f32_16x16x32_bf16(false, aH.v, false, bL.v, (short)0, c[nt].v, false, false);
      c[nt].v = __builtin_amdgcn_wmma_f32_16x16x32_bf16(false, aL.v, false, bH.v, (short)0, c[nt].v, false, false);
    }
  }
  // epilogue: proj -> split bf16, row-major [M, 224]
  #pragma unroll
  for (int nt = 0; nt < 14; ++nt) {
    #pragma unroll
    for (int r = 0; r < 8; ++r) {
      int mm = Mtile * 16 + half * 8 + r;      // C layout: reg r -> M = r + 8*half
      int nn = nt * 16 + l16;                  //           N = lane&15
      unsigned short h, l; split_bf16(c[nt].f[r], h, l);
      projH[(size_t)mm * N1P + nn] = h;
      projL[(size_t)mm * N1P + nn] = l;
    }
  }
}

// ---------------- GEMM2: scores[b] = proj[b] @ roles[b]^T  (A-register-stationary)
__global__ void __launch_bounds__(256)
gemm2_bf16x3(const unsigned short* __restrict__ projH, const unsigned short* __restrict__ projL,
             const unsigned short* __restrict__ rolH, const unsigned short* __restrict__ rolL,
             float* __restrict__ out) {
  const int lane = threadIdx.x & 31;
  const int wave = threadIdx.x >> 5;
  const int l16  = lane & 15;
  const int half = lane >> 4;
  const int Mtile = blockIdx.x * 8 + wave;     // 0..2047
  const int mrow  = Mtile * 16 + l16;
  const int b     = Mtile >> 6;                // 1024 rows per batch, 64 tiles per batch

  // Load ALL A fragments for K=224 once (112 VGPRs), reuse across 19 N-tiles.
  FragAB aH[7], aL[7];
  const unsigned short* aHrow = projH + (size_t)mrow * K2P + half * 8;
  const unsigned short* aLrow = projL + (size_t)mrow * K2P + half * 8;
  #pragma unroll
  for (int ks = 0; ks < 7; ++ks) {
    aH[ks].q[0] = *(const uint4*)(aHrow + ks * 32);
    aH[ks].q[1] = *(const uint4*)(aHrow + ks * 32 + 16);
    aL[ks].q[0] = *(const uint4*)(aLrow + ks * 32);
    aL[ks].q[1] = *(const uint4*)(aLrow + ks * 32 + 16);
  }
  const unsigned short* bHbase = rolH + (size_t)b * (N2P * K2P);
  const unsigned short* bLbase = rolL + (size_t)b * (N2P * K2P);

  for (int nt = 0; nt < 19; ++nt) {
    const unsigned short* bHrow = bHbase + (size_t)(nt * 16 + l16) * K2P + half * 8;
    const unsigned short* bLrow = bLbase + (size_t)(nt * 16 + l16) * K2P + half * 8;
    __builtin_prefetch(bHbase + (size_t)((nt + 1) * 16 + l16) * K2P, 0, 0);
    FragC c;
    #pragma unroll
    for (int r = 0; r < 8; ++r) c.f[r] = 0.0f;
    #pragma unroll
    for (int ks = 0; ks < 7; ++ks) {
      FragAB bH, bL;
      bH.q[0] = *(const uint4*)(bHrow + ks * 32);
      bH.q[1] = *(const uint4*)(bHrow + ks * 32 + 16);
      bL.q[0] = *(const uint4*)(bLrow + ks * 32);
      bL.q[1] = *(const uint4*)(bLrow + ks * 32 + 16);
      c.v = __builtin_amdgcn_wmma_f32_16x16x32_bf16(false, aH[ks].v, false, bH.v, (short)0, c.v, false, false);
      c.v = __builtin_amdgcn_wmma_f32_16x16x32_bf16(false, aH[ks].v, false, bL.v, (short)0, c.v, false, false);
      c.v = __builtin_amdgcn_wmma_f32_16x16x32_bf16(false, aL[ks].v, false, bH.v, (short)0, c.v, false, false);
    }
    #pragma unroll
    for (int r = 0; r < 8; ++r) {
      int nn = nt * 16 + l16;
      if (nn < Vz) {
        int mm = Mtile * 16 + half * 8 + r;
        out[(size_t)mm * Vz + nn] = (nn == 1) ? 0.0f : c.f[r];
      }
    }
  }
}

extern "C" void kernel_launch(void* const* d_in, const int* in_sizes, int n_in,
                              void* d_out, int out_size, void* d_ws, size_t ws_size,
                              hipStream_t stream) {
  (void)in_sizes; (void)n_in; (void)out_size; (void)ws_size;
  const float* role   = (const float*)d_in[0];  // [32, 300*200]
  const float* pre    = (const float*)d_in[1];  // [32, 1024, 300]
  const float* wid    = (const float*)d_in[2];  // [32, 1024, 16]
  const float* matrix = (const float*)d_in[3];  // [316, 200]
  float* out = (float*)d_out;                   // [32768, 300]

  // Workspace layout (split-bf16 operands; all 16B aligned)
  unsigned short* ws    = (unsigned short*)d_ws;
  unsigned short* projH = ws;                               // 32768*224
  unsigned short* projL = projH + (size_t)Mz * N1P;
  unsigned short* mTH   = projL + (size_t)Mz * N1P;         // 224*320
  unsigned short* mTL   = mTH + (size_t)N1P * K1P;
  unsigned short* rolH  = mTL + (size_t)N1P * K1P;          // 32*304*224
  unsigned short* rolL  = rolH + (size_t)Bz * N2P * K2P;

  prep_matT <<<(N1P * K1P + 255) / 256, 256, 0, stream>>>(matrix, mTH, mTL);
  prep_roles<<<(Bz * N2P * K2P + 255) / 256, 256, 0, stream>>>(role, rolH, rolL);
  gemm1_bf16x3<<<Mz / 16 / 8, 256, 0, stream>>>(pre, wid, mTH, mTL, projH, projL);
  gemm2_bf16x3<<<Mz / 16 / 8, 256, 0, stream>>>(projH, projL, rolH, rolL, out);
}
